// N3Aggregation2D_23665269801169
// MI455X (gfx1250) — compile-verified, matrix operand order checked by reading, not run on hardware
//
#include <hip/hip_runtime.h>
#include <hip/hip_bf16.h>

#ifndef __has_builtin
#define __has_builtin(x) 0
#endif

#define HP   66
#define QTOT 4356
#define BIGF 1e10f

typedef __attribute__((ext_vector_type(2)))  float          v2f;
typedef __attribute__((ext_vector_type(8)))  float          v8f;
typedef __attribute__((ext_vector_type(16))) __bf16         v16bf;
typedef __attribute__((ext_vector_type(16))) unsigned short v16us;
typedef __attribute__((ext_vector_type(4)))  unsigned       u32x4;
typedef __attribute__((ext_vector_type(8)))  unsigned       u32x8;

__device__ inline v8f vzero8() {
    v8f z = {0.f,0.f,0.f,0.f,0.f,0.f,0.f,0.f};
    return z;
}

// float -> bf16 with round-to-nearest-even
__device__ inline unsigned short f2bf(float f) {
    unsigned u = __builtin_bit_cast(unsigned, f);
    unsigned r = u + 0x7FFFu + ((u >> 16) & 1u);
    return (unsigned short)(r >> 16);
}

__device__ inline unsigned rfl(unsigned x) {
    return (unsigned)__builtin_amdgcn_readfirstlane((int)x);
}

// ---------------------------------------------------------------------------
// TDM gather: one descriptor moves 16 scattered 256-byte rows (16-bit data,
// row width 128 elements) from global into contiguous LDS. D# bitfields per
// CDNA5 ISA 8.3-8.7 (count=1, gather_mode=1, 16-bit indices, data_size=2B).
// Tracked by TENSORcnt.
// ---------------------------------------------------------------------------
__device__ inline void tdm_gather16(u32x4 g0, u32x8 g1, u32x4 g2, u32x4 g3) {
    asm volatile("tensor_load_to_lds %0, %1, %2, %3"
                 :: "s"(g0), "s"(g1), "s"(g2), "s"(g3)
                 : "memory");
}

__device__ inline void tdm_wait0() {
#if __has_builtin(__builtin_amdgcn_s_wait_tensorcnt)
    __builtin_amdgcn_s_wait_tensorcnt(0);
#else
    asm volatile("s_wait_tensorcnt 0x0" ::: "memory");
#endif
    asm volatile("" ::: "memory");
}

// ---------------------------------------------------------------------------
// Kernel 1: patch embeddings of xe/ye (Q x 64), their norms, exp(-mean(lt)),
// and bf16 x-patches (Q x 128). Query (i,j) lives on the padded 66x66 grid;
// patch element (a,b) reads original image at (i+a-3, j+b-3), zero outside.
// ---------------------------------------------------------------------------
__global__ __launch_bounds__(256) void k_prep(
    const float* __restrict__ x,  const float* __restrict__ xe,
    const float* __restrict__ ye, const float* __restrict__ lt,
    float* __restrict__ pex, float* __restrict__ pey,
    float* __restrict__ cn,  float* __restrict__ qn,
    float* __restrict__ tinv, unsigned short* __restrict__ pxh)
{
    int q = blockIdx.x * blockDim.x + threadIdx.x;
    if (q >= QTOT) return;
    int i = q / HP, j = q % HP;

    float cna = 0.f, qna = 0.f;
    #pragma unroll
    for (int e = 0; e < 4; e++)
        #pragma unroll
        for (int a = 0; a < 4; a++)
            #pragma unroll
            for (int b = 0; b < 4; b++) {
                int yy = i + a - 3, xx = j + b - 3;
                bool in = (yy >= 0 && yy < 64 && xx >= 0 && xx < 64);
                float vx = in ? xe[e*4096 + yy*64 + xx] : 0.f;
                float vy = in ? ye[e*4096 + yy*64 + xx] : 0.f;
                int d = e*16 + a*4 + b;
                pex[q*64 + d] = vx; cna += vx * vx;
                pey[q*64 + d] = vy; qna += vy * vy;
            }
    cn[q] = cna; qn[q] = qna;

    float s = 0.f;
    #pragma unroll
    for (int a = 0; a < 4; a++)
        #pragma unroll
        for (int b = 0; b < 4; b++) {
            int yy = i + a - 3, xx = j + b - 3;
            s += (yy >= 0 && yy < 64 && xx >= 0 && xx < 64) ? lt[yy*64 + xx] : 0.f;
        }
    tinv[q] = __expf(-s * (1.f / 16.f));

    #pragma unroll
    for (int ch = 0; ch < 8; ch++)
        #pragma unroll
        for (int a = 0; a < 4; a++)
            #pragma unroll
            for (int b = 0; b < 4; b++) {
                int yy = i + a - 3, xx = j + b - 3;
                float vx = (yy >= 0 && yy < 64 && xx >= 0 && xx < 64)
                         ? x[ch*4096 + yy*64 + xx] : 0.f;
                pxh[q*128 + ch*16 + a*4 + b] = f2bf(vx);
            }
}

// ---------------------------------------------------------------------------
// Kernel 2: per 16-query row tile:
//   phase 1 : 16x64x480 fp32 cross GEMM via v_wmma_f32_16x16x4_f32 -> LDS dists
//   phase 2a: per-query top-64 (wave argmin) + K=7 softmax relaxation weights
//   phase 2b: TDM-gather of selected patch rows into LDS, then
//             zp = Wk(7x64) @ gath(64x128) via v_wmma_f32_16x16x32_bf16
// ---------------------------------------------------------------------------
__global__ __launch_bounds__(128) void k_main(
    const float* __restrict__ pex, const float* __restrict__ pey,
    const float* __restrict__ cn,  const float* __restrict__ qn,
    const float* __restrict__ tinv, const unsigned short* __restrict__ pxh,
    float* __restrict__ zp)
{
    // LDS map (bytes):
    //   [0,32768)      : phase1/2a -> dist matrix 16x480 f32 (30720 used)
    //                    phase2b   -> 4 waves x 8KB bf16 gather chunks (alias)
    //   [32768,47104)  : Wk bf16, 16 queries x 7 x 64
    //   [47104,51200)  : selected neighbor indices, 16 x 64 int
    //   [51200,51328)  : qn / tinv for the 16 tile queries
    __shared__ __align__(16) char smem[51456];
    float*          distS  = (float*)smem;
    unsigned short* gch16  = (unsigned short*)smem;
    unsigned short* wkS    = (unsigned short*)(smem + 32768);
    int*            indsS  = (int*)(smem + 47104);
    float*          qnS    = (float*)(smem + 51200);
    float*          tinvS  = (float*)(smem + 51264);

    const int tid  = threadIdx.x;
    const int lane = tid & 31;
    const int wave = tid >> 5;
    const int nl   = lane & 15;
    const int hi   = (lane >= 16) ? 1 : 0;
    const int i    = blockIdx.y;
    const int j0   = blockIdx.x * 16;
    const int top      = min(max(i  - 7, 0), 51);
    const int baseleft = min(max(j0 - 7, 0), 51);

    if (tid < 16) {
        int jq = min(j0 + tid, HP - 1);
        qnS[tid]   = qn[i*HP + jq];
        tinvS[tid] = tinv[i*HP + jq];
    }
    __syncthreads();

    // ---------------- phase 1: fp32 WMMA cross distances --------------------
    {
        const int Mq = i*HP + min(j0 + nl, HP - 1);   // query row for A (M=nl)
        const int kb = hi ? 2 : 0;                    // fp32 A/B K striping
        const float* ap = pey + Mq*64 + kb;

        // A fragments and qn values are nt-invariant: hoist (8x reuse)
        v2f afr[16];
        #pragma unroll
        for (int ks = 0; ks < 16; ks++) {
            afr[ks].x = ap[ks*4]; afr[ks].y = ap[ks*4 + 1];
        }
        float qv[8];
        #pragma unroll
        for (int r8 = 0; r8 < 8; r8++) qv[r8] = qnS[r8 + (hi ? 8 : 0)];

        for (int nt = wave; nt < 30; nt += 4) {
            const int N  = nt*16 + nl;                // union-candidate index
            const int r  = N >> 5, cc = N & 31;
            const int candq = (top + r)*HP + min(baseleft + cc, HP - 1);
            v8f acc = vzero8();
            const float* bp = pex + candq*64 + kb;
            #pragma unroll
            for (int ks = 0; ks < 16; ks++) {
                v2f b;
                b.x = bp[ks*4]; b.y = bp[ks*4 + 1];
                acc = __builtin_amdgcn_wmma_f32_16x16x4_f32(
                          false, afr[ks], false, b, (short)0, acc, false, false);
            }
            const float cnv = cn[candq];
            #pragma unroll
            for (int r8 = 0; r8 < 8; r8++) {
                int M = r8 + (hi ? 8 : 0);            // C/D row layout
                distS[M*480 + N] = qv[r8] + cnv - 2.f * acc[r8];
            }
        }
    }
    __syncthreads();

    // ---------------- phase 2a: top-64 + neural-NN softmax ------------------
    #pragma unroll 1
    for (int t = 0; t < 4; t++) {
        const int lq = wave*4 + t;
        const int j  = j0 + lq;
        if (j >= HP) continue;
        const int q = i*HP + j;
        const int leftj = min(max(j - 7, 0), 51);
        const int shift = leftj - baseleft;

        float dv[8]; int ci[8];
        #pragma unroll
        for (int s = 0; s < 8; s++) {
            int o = s*32 + lane;                      // candidate ordinal
            if (o < 225) {
                int rr = o / 15, c2 = o - rr*15;
                float dd = distS[lq*480 + (rr << 5) + shift + c2];
                int cq = (top + rr)*HP + leftj + c2;
                if (cq == q) dd = BIGF;               // remove_self
                dv[s] = dd; ci[s] = cq;
            } else { dv[s] = BIGF; ci[s] = 0; }
        }

        float selD[2]; int selI[2];
        for (int t64 = 0; t64 < 64; t64++) {
            float bv = dv[0]; int bo = lane; int bc = ci[0];
            #pragma unroll
            for (int s = 1; s < 8; s++) {
                int o = s*32 + lane;
                if (dv[s] < bv) { bv = dv[s]; bo = o; bc = ci[s]; }
            }
            #pragma unroll
            for (int off = 16; off >= 1; off >>= 1) {
                float ov = __shfl_xor(bv, off, 32);
                int   oo = __shfl_xor(bo, off, 32);
                int   oc = __shfl_xor(bc, off, 32);
                if (ov < bv || (ov == bv && oo < bo)) { bv = ov; bo = oo; bc = oc; }
            }
            if (lane == (t64 & 31)) { selD[t64 >> 5] = bv; selI[t64 >> 5] = bc; }
            if (lane == (bo & 31)) {
                int ws2 = bo >> 5;
                #pragma unroll
                for (int s = 0; s < 8; s++) if (s == ws2) dv[s] = BIGF;
            }
        }
        indsS[lq*64 + lane]      = selI[0];
        indsS[lq*64 + 32 + lane] = selI[1];

        const float tv = tinvS[lq];
        float l0 = -selD[0] * tv, l1 = -selD[1] * tv;   // logits = -d / temp
        #pragma unroll 1
        for (int k = 0; k < 7; k++) {
            float m = fmaxf(l0, l1);
            #pragma unroll
            for (int off = 16; off >= 1; off >>= 1) m = fmaxf(m, __shfl_xor(m, off, 32));
            float e0 = __expf(l0 - m), e1 = __expf(l1 - m);
            float sm = e0 + e1;
            #pragma unroll
            for (int off = 16; off >= 1; off >>= 1) sm += __shfl_xor(sm, off, 32);
            float w0 = e0 / sm, w1 = e1 / sm;
            wkS[lq*448 + k*64 + lane]      = f2bf(w0);
            wkS[lq*448 + k*64 + 32 + lane] = f2bf(w1);
            l0 += __logf(fmaxf(1.f - w0, 1e-10f));
            l1 += __logf(fmaxf(1.f - w1, 1e-10f));
        }
    }
    __syncthreads();    // dist region now dead -> reuse as TDM gather buffers

    // ---------------- phase 2b: TDM gather + bf16 WMMA ----------------------
    unsigned short* gw16 = gch16 + wave*4096;   // per-wave 8KB chunk
    const unsigned gwLds =
        (unsigned)(uintptr_t)(void*)gw16;        // LDS byte offset of chunk
    const unsigned long long gbase = (unsigned long long)(uintptr_t)pxh;
    const unsigned glo = rfl((unsigned)gbase);
    const unsigned g3hi = rfl(((unsigned)(gbase >> 32) & 0x01FFFFFFu) |
                              (2u << 30));      // addr[56:32] | type=2

    // descriptor group1 (constant for all gathers):
    //   data_size=2B; tensor_dim0=128; tensor_dim1=QTOT; tile_dim0=128;
    //   tile_dim1=16 indices; tensor_dim0_stride=128
    const u32x8 g1c = { 0x00010000u,
                        (128u & 0xFFFFu) << 16,
                        ((unsigned)QTOT & 0xFFFFu) << 16,
                        128u << 16,
                        16u,
                        128u,
                        0u, 0u };

    #pragma unroll 1
    for (int t = 0; t < 4; t++) {
        const int lq = wave*4 + t;
        const int j  = j0 + lq;
        if (j >= HP) continue;
        const int q = i*HP + j;

        v8f acc[8];
        #pragma unroll
        for (int n = 0; n < 8; n++) acc[n] = vzero8();

        #pragma unroll 1
        for (int kk = 0; kk < 64; kk += 32) {
            // two TDM gather descriptors: 16 selected rows each -> LDS chunk
            #pragma unroll 1
            for (int h = 0; h < 2; h++) {
                unsigned p[8];
                #pragma unroll
                for (int n2 = 0; n2 < 8; n2++) {
                    unsigned r0 = (unsigned)indsS[lq*64 + kk + h*16 + 2*n2];
                    unsigned r1 = (unsigned)indsS[lq*64 + kk + h*16 + 2*n2 + 1];
                    p[n2] = rfl(r0 | (r1 << 16));
                }
                u32x4 g0 = { 0x80000001u,              // count=1, gather_mode
                             rfl(gwLds + (unsigned)h*4096u),
                             glo, g3hi };
                u32x4 g2 = { p[0], p[1], p[2], p[3] };
                u32x4 g3 = { p[4], p[5], p[6], p[7] };
                tdm_gather16(g0, g1c, g2, g3);
            }
            tdm_wait0();

            // A fragment: Wk rows 0..6 (zero-padded to 16), K = neighbor idx
            v16us au;
            #pragma unroll
            for (int e = 0; e < 16; e++) {
                int Kk = (e & 7) + ((e >> 3) << 4) + (hi ? 8 : 0);
                au[e] = (nl < 7) ? wkS[lq*448 + nl*64 + kk + Kk]
                                 : (unsigned short)0;
            }
            v16bf a = __builtin_bit_cast(v16bf, au);

            #pragma unroll
            for (int n = 0; n < 8; n++) {
                v16us bu;
                int N = n*16 + nl;                    // output dim 0..127
                #pragma unroll
                for (int e = 0; e < 16; e++) {
                    int Kk = (e & 7) + ((e >> 3) << 4) + (hi ? 8 : 0);
                    bu[e] = gw16[Kk*128 + N];
                }
                v16bf b = __builtin_bit_cast(v16bf, bu);
                acc[n] = __builtin_amdgcn_wmma_f32_16x16x32_bf16(
                             false, a, false, b, (short)0, acc[n], false, false);
            }
        }
        #pragma unroll
        for (int n = 0; n < 8; n++) {
            #pragma unroll
            for (int r8 = 0; r8 < 8; r8++) {
                int M = r8 + (hi ? 8 : 0);
                if (M < 7)
                    zp[(q*7 + M)*128 + n*16 + nl] = acc[n][r8];
            }
        }
    }
}

// ---------------------------------------------------------------------------
// Kernel 3: gather-based fold + normalize + residual + concat + crop.
// Output (1,64,64,64): ch<8 -> y passthrough; ch>=8 -> z[k,c] - y[c].
// ---------------------------------------------------------------------------
__global__ __launch_bounds__(256) void k_fold(
    const float* __restrict__ zp, const float* __restrict__ y,
    float* __restrict__ out)
{
    int idx = blockIdx.x * blockDim.x + threadIdx.x;
    if (idx >= 64*64*64) return;
    int ch  = idx >> 12;
    int pix = idx & 4095;
    int ii = pix >> 6, jj = pix & 63;
    if (ch < 8) { out[idx] = y[ch*4096 + pix]; return; }

    int k = (ch - 8) >> 3, c = (ch - 8) & 7;
    int pi = ii + 1, pj = jj + 1;                // padded-grid coordinates
    float sum = 0.f, cnt = 0.f;
    #pragma unroll
    for (int oa = -2; oa <= 1; oa++)
        #pragma unroll
        for (int ob = -2; ob <= 1; ob++) {
            int qi = pi - oa, qj = pj - ob;
            if (qi >= 0 && qi < HP && qj >= 0 && qj < HP) {
                int q = qi*HP + qj;
                int p = (oa + 2)*4 + (ob + 2);
                sum += zp[(q*7 + k)*128 + c*16 + p];
                cnt += 1.f;
            }
        }
    out[idx] = sum / (cnt + 1e-10f) - y[c*4096 + pix];
}

// ---------------------------------------------------------------------------
extern "C" void kernel_launch(void* const* d_in, const int* in_sizes, int n_in,
                              void* d_out, int out_size, void* d_ws, size_t ws_size,
                              hipStream_t stream)
{
    const float* x  = (const float*)d_in[0];   // (1,8,64,64)
    const float* xe = (const float*)d_in[1];   // (1,4,64,64)
    const float* ye = (const float*)d_in[2];   // (1,4,64,64)
    const float* y  = (const float*)d_in[3];   // (1,8,64,64)
    const float* lt = (const float*)d_in[4];   // (1,1,64,64)

    char* ws = (char*)d_ws;
    float*          pex  = (float*)(ws + 0);                // Q*64  f32
    float*          pey  = (float*)(ws + 1115136);          // Q*64  f32
    float*          cn   = (float*)(ws + 2230272);          // Q     f32
    float*          qn   = (float*)(ws + 2247936);          // Q     f32
    float*          tinv = (float*)(ws + 2265600);          // Q     f32
    unsigned short* pxh  = (unsigned short*)(ws + 2283264); // Q*128 bf16
    float*          zp   = (float*)(ws + 3398400);          // Q*7*128 f32
    float*          out  = (float*)d_out;

    k_prep<<<dim3(18), dim3(256), 0, stream>>>(x, xe, ye, lt,
                                               pex, pey, cn, qn, tinv, pxh);
    k_main<<<dim3(5, 66), dim3(128), 0, stream>>>(pex, pey, cn, qn, tinv,
                                                  pxh, zp);
    k_fold<<<dim3(1024), dim3(256), 0, stream>>>(zp, y, out);
}
